// TorchRPTAttention_5892695130606
// MI455X (gfx1250) — matrix operand order repaired and checked by reading, not compiled, not run on hardware
//
#include <hip/hip_runtime.h>
#include <math.h>

// ---------------------------------------------------------------------------
// Types
// ---------------------------------------------------------------------------
typedef __attribute__((ext_vector_type(8)))  __bf16 bf16x8;
typedef __attribute__((ext_vector_type(16))) __bf16 bf16x16;
typedef __attribute__((ext_vector_type(8)))  float  f32x8;

#define DEV __device__ __forceinline__

constexpr int B_   = 2;
constexpr int S_   = 2048;
constexpr int H_   = 4096;
constexpr int NH_  = 32;
constexpr int NKV_ = 8;
constexpr int D_   = 128;
constexpr int NREP_ = NH_ / NKV_;
constexpr int MROWS = B_ * S_;      // 4096 token rows

// ---------------------------------------------------------------------------
// bf16 helpers (bit-level, no libdevice dependency)
// ---------------------------------------------------------------------------
DEV __bf16 f2bf(float f) {
  unsigned u = __builtin_bit_cast(unsigned, f);
  u += 0x7fffu + ((u >> 16) & 1u);                 // round-to-nearest-even
  unsigned short h = (unsigned short)(u >> 16);
  return __builtin_bit_cast(__bf16, h);
}
DEV float bf2f(__bf16 b) {
  unsigned short h = __builtin_bit_cast(unsigned short, b);
  unsigned u = ((unsigned)h) << 16;
  return __builtin_bit_cast(float, u);
}
DEV bf16x16 cat8(bf16x8 lo, bf16x8 hi) {
  return __builtin_shufflevector(lo, hi, 0,1,2,3,4,5,6,7,8,9,10,11,12,13,14,15);
}

// A fragment 16x32 bf16 (CDNA5 layout):
//   lanes 0-15 : M=lane,   K chunks {0..7, 16..23}
//   lanes16-31 : M=lane-16,K chunks {8..15,24..31}
DEV bf16x16 load_a_frag(const __bf16* base, int stride, int lane) {
  int m  = lane & 15;
  int c0 = (lane >> 4) << 3;                       // 0 or 8
  const __bf16* p = base + (long)m * stride + c0;
  return cat8(*(const bf16x8*)p, *(const bf16x8*)(p + 16));
}
// B fragment 32x16 bf16 (CDNA5 layout):
//   lanes 0-15 : N=lane,   K chunks {0..7, 16..23}
//   lanes16-31 : N=lane-16,K chunks {8..15,24..31}
DEV bf16x16 load_b_frag(const __bf16* base, int stride, int lane) {
  int n  = lane & 15;
  int k0 = (lane >> 4) << 3;                       // 0 or 8
  const __bf16* p = base + (long)n * stride + k0;
  return cat8(*(const bf16x8*)p, *(const bf16x8*)(p + 16));
}

DEV f32x8 wmma_bf16(bf16x16 a, bf16x16 b, f32x8 c) {
  return __builtin_amdgcn_wmma_f32_16x16x32_bf16(false, a, false, b,
                                                 (short)0, c, false, false);
}

// ---------------------------------------------------------------------------
// Async copy (gfx1250): global -> LDS without touching VGPR data,
// tracked by ASYNCcnt. GV mode: vdst = LDS byte offset, vaddr = 64-bit addr.
// ---------------------------------------------------------------------------
DEV void async_copy_b128(const void* gptr, void* lds_ptr) {
  unsigned long long ga = (unsigned long long)gptr;
  unsigned la = (unsigned)(uintptr_t)lds_ptr;      // low 32 bits = LDS offset
  asm volatile("global_load_async_to_lds_b128 %0, %1, off"
               :: "v"(la), "v"(ga) : "memory");
}
DEV void wait_async0() {
  asm volatile("s_wait_asynccnt 0" ::: "memory");
}

// ---------------------------------------------------------------------------
// Elementwise conversion / transpose kernels
// ---------------------------------------------------------------------------
__global__ void f32_to_bf16_kernel(const float* __restrict__ in,
                                   __bf16* __restrict__ out, long n) {
  long i = (long)blockIdx.x * blockDim.x + threadIdx.x;
  if (i < n) out[i] = f2bf(in[i]);
}

// w[K,N] (f32, row-major) -> wt[N,K] (bf16, row-major)
__global__ void wtrans_kernel(const float* __restrict__ w,
                              __bf16* __restrict__ wt, int K, int N) {
  long i = (long)blockIdx.x * blockDim.x + threadIdx.x;
  long total = (long)K * N;
  if (i >= total) return;
  int k = (int)(i % K);
  int n = (int)(i / K);
  wt[i] = f2bf(w[(long)k * N + n]);
}

// v[B,S,NKV,D] bf16 -> vt[B,NKV,D,S] bf16
__global__ void vtrans_kernel(const __bf16* __restrict__ v,
                              __bf16* __restrict__ vt) {
  long i = (long)blockIdx.x * blockDim.x + threadIdx.x;
  long total = (long)B_ * NKV_ * D_ * S_;
  if (i >= total) return;
  int s = (int)(i % S_);  long t = i / S_;
  int d = (int)(t % D_);  t /= D_;
  int h = (int)(t % NKV_);
  int b = (int)(t / NKV_);
  vt[i] = v[(((long)(b * S_ + s)) * NKV_ + h) * D_ + d];
}

// RoPE in-place on bf16 [B,S,heads,D]
__global__ void rope_kernel(__bf16* __restrict__ x, int heads) {
  long i = (long)blockIdx.x * blockDim.x + threadIdx.x;
  long total = (long)B_ * S_ * heads * (D_ / 2);
  if (i >= total) return;
  int p = (int)(i % (D_ / 2));  long t = i / (D_ / 2);
  int h = (int)(t % heads);     t /= heads;
  int s = (int)(t % S_);
  int b = (int)(t / S_);
  // inv_freq = theta^(-2p/D),  ln(10000) = 9.210340372
  float inv = __expf(-(2.0f * (float)p / (float)D_) * 9.210340371976184f);
  float ang = (float)s * inv;
  float c = __cosf(ang), sn = __sinf(ang);
  __bf16* ptr = x + ((((long)(b * S_ + s)) * heads + h) * D_ + 2 * p);
  float xr = bf2f(ptr[0]), xi = bf2f(ptr[1]);
  ptr[0] = f2bf(xr * c - xi * sn);
  ptr[1] = f2bf(xr * sn + xi * c);
}

// ---------------------------------------------------------------------------
// bf16 WMMA GEMM with async double-buffered LDS staging:
//   C[M,N] = A[M,K] @ Bt[N,K]^T      (Bt pre-transposed, so both tiles are
//                                     K-contiguous 16B-chunk copies)
// Block: 256 threads = 8 waves (4M x 2N); wave tile 32x64 (2x4 WMMAs).
// Block tile 128(M) x 128(N); K stepped by 32.
// LDS: 2 x (128x32 A + 128x32 B) bf16 = 32 KB, double buffered.
// Fill path: global_load_async_to_lds_b128 (ASYNCcnt), fenced with
// s_wait_asynccnt 0 + workgroup barrier; compute reads via ds_load_b128.
// ---------------------------------------------------------------------------
template <typename OutT>
__global__ __launch_bounds__(256) void gemm_wmma(const __bf16* __restrict__ A,
                                                 const __bf16* __restrict__ Bt,
                                                 OutT* __restrict__ C,
                                                 int M, int N, int K) {
  __shared__ __align__(16) __bf16 ldsA[2][128][32];
  __shared__ __align__(16) __bf16 ldsB[2][128][32];

  const int t    = threadIdx.x;
  const int lane = t & 31;
  const int wid  = t >> 5;
  const int mb0  = blockIdx.y * 128;
  const int nb0  = blockIdx.x * 128;
  const int wm   = (wid & 3) * 32;    // wave M offset within block tile
  const int wn   = (wid >> 2) * 64;   // wave N offset within block tile

  // Copy assignment: thread t moves one 32B run (two b128) of A and of B.
  const int crow = t >> 1;            // 0..127
  const int ccol = (t & 1) * 16;      // 0 or 16 (elements)

  const __bf16* Ag = A  + (long)(mb0 + crow) * K + ccol;
  const __bf16* Bg = Bt + (long)(nb0 + crow) * K + ccol;

  // prologue: fill buffer 0 with k0 = 0
  async_copy_b128(Ag,     &ldsA[0][crow][ccol]);
  async_copy_b128(Ag + 8, &ldsA[0][crow][ccol + 8]);
  async_copy_b128(Bg,     &ldsB[0][crow][ccol]);
  async_copy_b128(Bg + 8, &ldsB[0][crow][ccol + 8]);
  wait_async0();
  __syncthreads();

  f32x8 acc[2][4] = {};
  int cur = 0;

  for (int k0 = 0; k0 < K; k0 += 32) {
    const int nxt = cur ^ 1;
    const bool more = (k0 + 32) < K;
    if (more) {  // kick off next K-slab into the other buffer
      async_copy_b128(Ag + k0 + 32,     &ldsA[nxt][crow][ccol]);
      async_copy_b128(Ag + k0 + 40,     &ldsA[nxt][crow][ccol + 8]);
      async_copy_b128(Bg + k0 + 32,     &ldsB[nxt][crow][ccol]);
      async_copy_b128(Bg + k0 + 40,     &ldsB[nxt][crow][ccol + 8]);
    }

    bf16x16 a[2];
    a[0] = load_a_frag(&ldsA[cur][wm][0],      32, lane);
    a[1] = load_a_frag(&ldsA[cur][wm + 16][0], 32, lane);
    bf16x16 b[4];
#pragma unroll
    for (int j = 0; j < 4; ++j)
      b[j] = load_b_frag(&ldsB[cur][wn + j * 16][0], 32, lane);
#pragma unroll
    for (int i = 0; i < 2; ++i)
#pragma unroll
      for (int j = 0; j < 4; ++j)
        acc[i][j] = wmma_bf16(a[i], b[j], acc[i][j]);

    if (more) {
      wait_async0();     // own async writes done
      __syncthreads();   // everyone's writes visible, reads of cur complete
    }
    cur = nxt;
  }

  // C layout: VGPR r -> (M = r + 8*(lane>=16), N = lane&15)
  const int nl = lane & 15;
  const int half8 = (lane >> 4) << 3;
#pragma unroll
  for (int i = 0; i < 2; ++i)
#pragma unroll
    for (int j = 0; j < 4; ++j)
#pragma unroll
      for (int r = 0; r < 8; ++r) {
        int row = mb0 + wm + i * 16 + r + half8;
        int col = nb0 + wn + j * 16 + nl;
        float v = acc[i][j][r];
        if constexpr (sizeof(OutT) == 2)
          C[(long)row * N + col] = f2bf(v);
        else
          C[(long)row * N + col] = v;
      }
}

// ---------------------------------------------------------------------------
// Flash attention: one wave per 16 query rows; 4 waves / block.
// Q (16x128) held in registers as 4 A-fragments; keys streamed 32 at a time.
// P converted C-layout -> A-layout through a per-wave LDS slab.
// V pre-transposed to [D,S] so PV B-fragments are contiguous.
// ---------------------------------------------------------------------------
__global__ __launch_bounds__(128) void flash_kernel(const __bf16* __restrict__ q,
                                                    const __bf16* __restrict__ k,
                                                    const __bf16* __restrict__ vt,
                                                    __bf16* __restrict__ o) {
  __shared__ __align__(16) __bf16 ldsP[4][16][32];

  const int lane = threadIdx.x & 31;
  const int wid  = threadIdx.x >> 5;
  const int qb = (blockIdx.x * 4 + wid) * 16;      // query-tile base
  const int h  = blockIdx.y;
  const int b  = blockIdx.z;
  const int kvh = h / NREP_;
  const int nl = lane & 15;
  const int half8 = (lane >> 4) << 3;
  const float scale = 0.08838834764831845f;        // 1/sqrt(128)

  const __bf16* qbase = q + (((long)(b * S_ + qb)) * NH_ + h) * D_;
  bf16x16 qf[4];
#pragma unroll
  for (int dstep = 0; dstep < 4; ++dstep)
    qf[dstep] = load_a_frag(qbase + dstep * 32, NH_ * D_, lane);

  f32x8 oacc[8] = {};
  float mrow[8], lrow[8];
#pragma unroll
  for (int r = 0; r < 8; ++r) { mrow[r] = -1e30f; lrow[r] = 0.0f; }

  const __bf16* kbase0 = k + (((long)(b * S_)) * NKV_ + kvh) * D_;
  const int kstride = NKV_ * D_;
  const __bf16* vtbase = vt + ((long)(b * NKV_ + kvh)) * D_ * S_;

  for (int kb = 0; kb < qb + 16; kb += 32) {       // causal streaming
    float sv[2][8];
#pragma unroll
    for (int kh = 0; kh < 2; ++kh) {
      f32x8 sacc = {};
#pragma unroll
      for (int dstep = 0; dstep < 4; ++dstep) {
        bf16x16 kf = load_b_frag(
            kbase0 + (long)(kb + kh * 16) * kstride + dstep * 32,
            kstride, lane);
        sacc = wmma_bf16(qf[dstep], kf, sacc);
      }
#pragma unroll
      for (int r = 0; r < 8; ++r) {
        int key  = kb + kh * 16 + nl;
        int qrow = qb + r + half8;
        sv[kh][r] = (key <= qrow) ? sacc[r] * scale : -1e30f;
      }
    }
    // online softmax per row (rows live in 16-lane halves)
#pragma unroll
    for (int r = 0; r < 8; ++r) {
      float tmax = fmaxf(sv[0][r], sv[1][r]);
#pragma unroll
      for (int mm = 1; mm < 16; mm <<= 1)
        tmax = fmaxf(tmax, __shfl_xor(tmax, mm, 32));
      float mnew  = fmaxf(mrow[r], tmax);
      float alpha = __expf(mrow[r] - mnew);
      float p0 = __expf(sv[0][r] - mnew);
      float p1 = __expf(sv[1][r] - mnew);
      float rs = p0 + p1;
#pragma unroll
      for (int mm = 1; mm < 16; mm <<= 1)
        rs += __shfl_xor(rs, mm, 32);
      lrow[r] = lrow[r] * alpha + rs;
      mrow[r] = mnew;
#pragma unroll
      for (int dc = 0; dc < 8; ++dc) oacc[dc][r] *= alpha;
      int row = r + half8;
      ldsP[wid][row][nl]      = f2bf(p0);          // C-layout -> LDS row-major
      ldsP[wid][row][16 + nl] = f2bf(p1);
    }
    // P back out of LDS in A-layout, then PV
    bf16x16 pf = load_a_frag(&ldsP[wid][0][0], 32, lane);
#pragma unroll
    for (int dc = 0; dc < 8; ++dc) {
      bf16x16 vf = load_b_frag(vtbase + (long)(dc * 16) * S_ + kb, S_, lane);
      oacc[dc] = wmma_bf16(pf, vf, oacc[dc]);
    }
  }

  // epilogue: normalize and store bf16 [B,S,NH*D]
#pragma unroll
  for (int r = 0; r < 8; ++r) {
    float invl = 1.0f / lrow[r];
    int row = qb + r + half8;
    __bf16* op = o + ((long)(b * S_ + row)) * (NH_ * D_) + h * D_;
#pragma unroll
    for (int dc = 0; dc < 8; ++dc)
      op[dc * 16 + nl] = f2bf(oacc[dc][r] * invl);
  }
}

// ---------------------------------------------------------------------------
// Host-side orchestration
// ---------------------------------------------------------------------------
extern "C" void kernel_launch(void* const* d_in, const int* in_sizes, int n_in,
                              void* d_out, int out_size, void* d_ws, size_t ws_size,
                              hipStream_t stream) {
  const float* x  = (const float*)d_in[0];
  const float* wq = (const float*)d_in[1];
  const float* wk = (const float*)d_in[2];
  const float* wv = (const float*)d_in[3];
  const float* wo = (const float*)d_in[4];
  float* out = (float*)d_out;

  char* p = (char*)d_ws;
  auto alloc = [&](size_t bytes) -> char* {
    char* r = p;
    p += (bytes + 255) & ~(size_t)255;
    return r;
  };
  __bf16* xb   = (__bf16*)alloc((size_t)MROWS * H_ * 2);
  __bf16* wqt  = (__bf16*)alloc((size_t)(NH_ * D_) * H_ * 2);
  __bf16* wkt  = (__bf16*)alloc((size_t)(NKV_ * D_) * H_ * 2);
  __bf16* wvt  = (__bf16*)alloc((size_t)(NKV_ * D_) * H_ * 2);
  __bf16* wot  = (__bf16*)alloc((size_t)H_ * (NH_ * D_) * 2);
  __bf16* qbuf = (__bf16*)alloc((size_t)MROWS * NH_ * D_ * 2);
  __bf16* kbuf = (__bf16*)alloc((size_t)MROWS * NKV_ * D_ * 2);
  __bf16* vbuf = (__bf16*)alloc((size_t)MROWS * NKV_ * D_ * 2);
  __bf16* vtb  = (__bf16*)alloc((size_t)MROWS * NKV_ * D_ * 2);
  __bf16* attn = (__bf16*)alloc((size_t)MROWS * NH_ * D_ * 2);

  // 1) activations -> bf16
  {
    long n = (long)MROWS * H_;
    f32_to_bf16_kernel<<<dim3((unsigned)((n + 255) / 256)), 256, 0, stream>>>(x, xb, n);
  }
  // 2) weights -> bf16, transposed [N,K]
  {
    long n = (long)(NH_ * D_) * H_;
    wtrans_kernel<<<dim3((unsigned)((n + 255) / 256)), 256, 0, stream>>>(wq, wqt, H_, NH_ * D_);
  }
  {
    long n = (long)(NKV_ * D_) * H_;
    wtrans_kernel<<<dim3((unsigned)((n + 255) / 256)), 256, 0, stream>>>(wk, wkt, H_, NKV_ * D_);
    wtrans_kernel<<<dim3((unsigned)((n + 255) / 256)), 256, 0, stream>>>(wv, wvt, H_, NKV_ * D_);
  }
  {
    long n = (long)H_ * (NH_ * D_);
    wtrans_kernel<<<dim3((unsigned)((n + 255) / 256)), 256, 0, stream>>>(wo, wot, NH_ * D_, H_);
  }

  // 3) QKV projections (bf16 WMMA GEMM, bf16 output)
  gemm_wmma<__bf16><<<dim3((NH_ * D_) / 128, MROWS / 128), 256, 0, stream>>>(
      xb, wqt, qbuf, MROWS, NH_ * D_, H_);
  gemm_wmma<__bf16><<<dim3((NKV_ * D_) / 128, MROWS / 128), 256, 0, stream>>>(
      xb, wkt, kbuf, MROWS, NKV_ * D_, H_);
  gemm_wmma<__bf16><<<dim3((NKV_ * D_) / 128, MROWS / 128), 256, 0, stream>>>(
      xb, wvt, vbuf, MROWS, NKV_ * D_, H_);

  // 4) RoPE on Q and K
  {
    long n = (long)B_ * S_ * NH_ * (D_ / 2);
    rope_kernel<<<dim3((unsigned)((n + 255) / 256)), 256, 0, stream>>>(qbuf, NH_);
  }
  {
    long n = (long)B_ * S_ * NKV_ * (D_ / 2);
    rope_kernel<<<dim3((unsigned)((n + 255) / 256)), 256, 0, stream>>>(kbuf, NKV_);
  }

  // 5) V -> [B,NKV,D,S] for contiguous PV B-fragments
  {
    long n = (long)B_ * NKV_ * D_ * S_;
    vtrans_kernel<<<dim3((unsigned)((n + 255) / 256)), 256, 0, stream>>>(vbuf, vtb);
  }

  // 6) Flash attention (WMMA QK^T + PV, online softmax)
  flash_kernel<<<dim3(S_ / 64, NH_, B_), 128, 0, stream>>>(qbuf, kbuf, vtb, attn);

  // 7) Output projection (f32 output to d_out)
  gemm_wmma<float><<<dim3(H_ / 128, MROWS / 128), 256, 0, stream>>>(
      attn, wot, out, MROWS, H_, NH_ * D_);
}
